// DynamicSpatialGNN_38903813767334
// MI455X (gfx1250) — matrix operand order repaired and checked by reading, not compiled, not run on hardware
//
#include <hip/hip_runtime.h>
#include <hip/hip_bf16.h>

typedef __attribute__((ext_vector_type(16))) _Float16 v16h;
typedef __attribute__((ext_vector_type(8)))  float    v8f;
typedef __attribute__((ext_vector_type(4)))  _Float16 h4;
typedef __attribute__((ext_vector_type(2)))  _Float16 h2;

union Frag { v16h v; h2 p[8]; };

#define WMMA16(a, b, c) __builtin_amdgcn_wmma_f32_16x16x32_f16(false, (a), false, (b), (short)0, (c), false, false)

// problem constants
constexpr int Nn = 64, Tt = 32, Dd = 128, HID = 32;

// LDS strides (in elements), padded to break 64-bank conflicts
constexpr int SH  = 132;  // f16, 128-wide row-major mats (zh/qh/kh/sp_h/wbuf)
constexpr int SVT = 68;   // f16, 64-wide (vT rows, alpha rows)
constexpr int SL  = 66;   // f32 logits
constexpr int SP  = 36;   // f32 qproj/kproj
constexpr int SR  = 132;  // f32 residual buffer

// LDS byte offsets
constexpr int OFF_ZH  = 0;
constexpr int OFF_QH  = OFF_ZH  + 64  * SH  * 2;
constexpr int OFF_KH  = OFF_QH  + 64  * SH  * 2;
constexpr int OFF_VT  = OFF_KH  + 64  * SH  * 2;
constexpr int OFF_WB  = OFF_VT  + 128 * SVT * 2;
constexpr int OFF_SPH = OFF_WB  + 128 * SH  * 2;
constexpr int OFF_AH  = OFF_SPH + 64  * SH  * 2;
constexpr int OFF_LG  = OFF_AH  + 64  * SVT * 2;
constexpr int OFF_QP  = OFF_LG  + 64  * SL  * 4;
constexpr int OFF_KP  = OFF_QP  + 64  * SP  * 4;
constexpr int OFF_RB  = OFF_KP  + 64  * SP  * 4;
constexpr int OFF_SM  = OFF_RB  + 64  * SR  * 4;
constexpr size_t SMEM_BYTES = (size_t)OFF_SM + 4800;  // ~201 KB of the WGP's 320 KB LDS

// ---- fragment loaders, contiguous k order k = 16*(l>>4) + s, same for A and B ----
// A (16x32 f16): lane l -> row M = l&15; VGPR v holds K pair:
//   k = (l>>4)*16 + 2v
__device__ __forceinline__ v16h load_a(const _Float16* src, int row0, int k0, int stride, int lane) {
    int m = lane & 15, hf = lane >> 4;
    Frag f;
#pragma unroll
    for (int v = 0; v < 8; ++v) {
        int k = k0 + (hf << 4) + (v << 1);
        f.p[v] = *(const h2*)(src + (row0 + m) * stride + k);
    }
    return f.v;
}
// B (32x16 f16): lane l -> col N = l&15; VGPR v holds K pair: k = (l>>4)*16 + 2v.
// src is indexed src[colRow][k] (i.e. pairs contiguous along K).
__device__ __forceinline__ v16h load_b(const _Float16* src, int col0, int k0, int stride, int lane) {
    int c = lane & 15, hf = lane >> 4;
    Frag f;
#pragma unroll
    for (int v = 0; v < 8; ++v) {
        int k = k0 + (hf << 4) + (v << 1);
        f.p[v] = *(const h2*)(src + (col0 + c) * stride + k);
    }
    return f.v;
}
// B fragment gathered from fp32 global memory with on-the-fly f16 convert
__device__ __forceinline__ v16h load_b_f32(const float* src, int col0, int k0, int stride, int lane) {
    int c = lane & 15, hf = lane >> 4;
    Frag f;
#pragma unroll
    for (int v = 0; v < 8; ++v) {
        int k = k0 + (hf << 4) + (v << 1);
        const float* p = src + (col0 + c) * stride + k;
        h2 t; t[0] = (_Float16)p[0]; t[1] = (_Float16)p[1];
        f.p[v] = t;
    }
    return f.v;
}

// 64xJ = (64xK) @ (KxJ), J = 128 (8 col tiles), 8 waves.
// Wave w: it = w&3, jt = (w>>2) + 2c, c = 0..3. K-outer loop: each A fragment
// is loaded once and reused for 4 WMMAs (4 live accumulators).
template<int KT, bool TR>
__device__ __forceinline__ void gemm_f16out(const _Float16* A, int sA,
                                            const _Float16* Bm, int sB,
                                            _Float16* dst, int sD,
                                            int w, int lane)
{
    const int it = w & 3, jb = w >> 2;
    const int col = lane & 15, hf = lane >> 4;
    v8f acc[4] = {};
#pragma unroll
    for (int kk = 0; kk < KT; ++kk) {
        v16h a = load_a(A, it * 16, kk * 32, sA, lane);
#pragma unroll
        for (int c = 0; c < 4; ++c) {
            v16h b = load_b(Bm, (jb + 2 * c) * 16, kk * 32, sB, lane);
            acc[c] = WMMA16(a, b, acc[c]);
        }
    }
#pragma unroll
    for (int c = 0; c < 4; ++c) {
        const int jt = jb + 2 * c;
#pragma unroll
        for (int r = 0; r < 8; ++r) {
            if (TR)
                dst[(jt * 16 + col) * sD + it * 16 + r + hf * 8] = (_Float16)acc[c][r];
            else
                dst[(it * 16 + r + hf * 8) * sD + jt * 16 + col] = (_Float16)acc[c][r];
        }
    }
}

// Same tiling, fp32 output (for the final Theta GEMM)
__device__ __forceinline__ void gemm_f32out(const _Float16* A, int sA,
                                            const _Float16* Bm, int sB,
                                            float* dst, int sD, int w, int lane)
{
    const int it = w & 3, jb = w >> 2;
    const int col = lane & 15, hf = lane >> 4;
    v8f acc[4] = {};
#pragma unroll
    for (int kk = 0; kk < 4; ++kk) {
        v16h a = load_a(A, it * 16, kk * 32, sA, lane);
#pragma unroll
        for (int c = 0; c < 4; ++c) {
            v16h b = load_b(Bm, (jb + 2 * c) * 16, kk * 32, sB, lane);
            acc[c] = WMMA16(a, b, acc[c]);
        }
    }
#pragma unroll
    for (int c = 0; c < 4; ++c) {
        const int jt = jb + 2 * c;
#pragma unroll
        for (int r = 0; r < 8; ++r)
            dst[(it * 16 + r + hf * 8) * sD + jt * 16 + col] = acc[c][r];
    }
}

// stage a 128x128 fp32 matrix (global, row-major) into f16 LDS, stride SH
__device__ __forceinline__ void stage_w(const float* __restrict__ src,
                                        _Float16* dst, int tid)
{
    for (int i4 = tid; i4 < (128 * 128) / 4; i4 += 256) {
        const int base = i4 * 4;
        const int r = base >> 7, c = base & 127;
        const float4 v = *(const float4*)(src + base);
        h4 o; o[0] = (_Float16)v.x; o[1] = (_Float16)v.y;
              o[2] = (_Float16)v.z; o[3] = (_Float16)v.w;
        *(h4*)(dst + r * SH + c) = o;
    }
}

__global__ __launch_bounds__(256) void dyn_spatial_gnn_kernel(
    const float* __restrict__ x, const float* __restrict__ edgef,
    const float* __restrict__ apri, const unsigned char* __restrict__ maskp,
    const float* __restrict__ Wq, const float* __restrict__ Wk,
    const float* __restrict__ Wv, const float* __restrict__ Theta,
    const float* __restrict__ Wfuse, const float* __restrict__ eW1,
    const float* __restrict__ eb1, const float* __restrict__ eW2,
    const float* __restrict__ eb2, const float* __restrict__ lng,
    const float* __restrict__ lnb, const float* __restrict__ physw_p,
    const float* __restrict__ priorw_p, float* __restrict__ out)
{
    extern __shared__ char smem[];
    _Float16* zh   = (_Float16*)(smem + OFF_ZH);
    _Float16* qh   = (_Float16*)(smem + OFF_QH);
    _Float16* kh   = (_Float16*)(smem + OFF_KH);
    _Float16* vT   = (_Float16*)(smem + OFF_VT);   // [d][m]
    _Float16* wbuf = (_Float16*)(smem + OFF_WB);
    _Float16* sph  = (_Float16*)(smem + OFF_SPH);
    _Float16* ah   = (_Float16*)(smem + OFF_AH);   // alpha f16 [n][m]
    float*    lg   = (float*)(smem + OFF_LG);
    float*    qp   = (float*)(smem + OFF_QP);
    float*    kp   = (float*)(smem + OFF_KP);
    float*    rb   = (float*)(smem + OFF_RB);
    float*    smf  = (float*)(smem + OFF_SM);
    float* gs    = smf;            // 128
    float* bs    = smf + 128;      // 128
    float* w1es  = smf + 256;      // 128 : W1e[h][e] at h*4+e
    float* eb1s  = smf + 384;      // 32
    float* ew2s  = smf + 416;      // 32
    float* wfs   = smf + 448;      // 8
    float* stats = smf + 456;      // 128 : mu/rstd interleaved
    float* part  = smf + 584;      // 512 : partial sums
    int*   maski = (int*)(smf + 1096); // 64

    const int tid  = threadIdx.x;
    const int lane = tid & 31;     // wave32
    const int w    = tid >> 5;     // 8 waves
    const int bt   = blockIdx.x;
    const int bi   = bt >> 5;      // bt = b*T + t
    const int ti   = bt & 31;
    const int col  = lane & 15, hf = lane >> 4;

    // ---- preload small tensors ----
    if (tid < 128) {
        gs[tid] = lng[tid];
        bs[tid] = lnb[tid];
        w1es[tid] = eW1[(tid >> 2) * 260 + 256 + (tid & 3)];
    }
    if (tid < 32) { eb1s[tid] = eb1[tid]; ew2s[tid] = eW2[tid]; }
    if (tid < 5)  wfs[tid] = Wfuse[tid];
    if (tid < 64) maski[tid] = maskp[bi * Nn + tid] ? 1 : 0;

    // ---- load Z (64x128) as f16, vectorized ----
    for (int i4 = tid; i4 < (Nn * Dd) / 4; i4 += 256) {
        const int base = i4 * 4;
        const int n = base >> 7, d = base & 127;
        const float4 v = *(const float4*)(x + (((bi * Nn + n) * Tt) + ti) * Dd + d);
        h4 o; o[0] = (_Float16)v.x; o[1] = (_Float16)v.y;
              o[2] = (_Float16)v.z; o[3] = (_Float16)v.w;
        *(h4*)(zh + n * SH + d) = o;
    }
    __syncthreads();

    // ---- Q = Z Wq^T ----
    stage_w(Wq, wbuf, tid);
    __syncthreads();
    gemm_f16out<4, false>(zh, SH, wbuf, SH, qh, SH, w, lane);
    __syncthreads();
    // ---- K = Z Wk^T ----
    stage_w(Wk, wbuf, tid);
    __syncthreads();
    gemm_f16out<4, false>(zh, SH, wbuf, SH, kh, SH, w, lane);
    __syncthreads();
    // ---- V = Z Wv^T, stored transposed [d][m] ----
    stage_w(Wv, wbuf, tid);
    __syncthreads();
    gemm_f16out<4, true>(zh, SH, wbuf, SH, vT, SVT, w, lane);
    __syncthreads();

    // ---- content = Q K^T / sqrt(D) : 4x4 tiles, 2 per wave ----
    {
        const int it = w & 3, jb = (w >> 2) * 2;
        const float scale = 0.08838834764831843f;  // 1/sqrt(128)
        v8f acc[2] = {};
#pragma unroll
        for (int kk = 0; kk < 4; ++kk) {
            v16h a = load_a(qh, it * 16, kk * 32, SH, lane);
#pragma unroll
            for (int c = 0; c < 2; ++c) {
                v16h b = load_b(kh, (jb + c) * 16, kk * 32, SH, lane);  // B[k=d][m]=K[m][d]
                acc[c] = WMMA16(a, b, acc[c]);
            }
        }
#pragma unroll
        for (int c = 0; c < 2; ++c)
#pragma unroll
            for (int r = 0; r < 8; ++r)
                lg[(it * 16 + r + hf * 8) * SL + (jb + c) * 16 + col] = acc[c][r] * scale;
    }

    // ---- qproj = Q W1q^T + eb1 ; kproj = K W1k^T (B gathered from global eW1) ----
    {
        const int it = w & 3, jt = w >> 2;  // jt in 0..1 (HID=32)
        v8f accq = {}, acck = {};
#pragma unroll
        for (int kk = 0; kk < 4; ++kk) {
            v16h bq = load_b_f32(eW1, jt * 16, kk * 32, 260, lane);
            v16h aq = load_a(qh, it * 16, kk * 32, SH, lane);
            accq = WMMA16(aq, bq, accq);
            v16h bk = load_b_f32(eW1 + 128, jt * 16, kk * 32, 260, lane);
            v16h ak = load_a(kh, it * 16, kk * 32, SH, lane);
            acck = WMMA16(ak, bk, acck);
        }
#pragma unroll
        for (int r = 0; r < 8; ++r) {
            qp[(it * 16 + r + hf * 8) * SP + jt * 16 + col] = accq[r] + eb1s[jt * 16 + col];
            kp[(it * 16 + r + hf * 8) * SP + jt * 16 + col] = acck[r];
        }
    }
    __syncthreads();

    // ---- edge MLP + prior + mask -> logits ----
    {
        const float physw = physw_p[0], priorw = priorw_p[0], eb2v = eb2[0];
        for (int idx = tid; idx < Nn * Nn; idx += 256) {
            const int n = idx >> 6, m = idx & 63;
            const float4 ef = *(const float4*)(edgef + ((size_t)bt * Nn * Nn + idx) * 4);
            float acc = 0.f;
#pragma unroll
            for (int h = 0; h < HID; ++h) {
                float hv = qp[n * SP + h] + kp[m * SP + h]
                         + w1es[h * 4 + 0] * ef.x + w1es[h * 4 + 1] * ef.y
                         + w1es[h * 4 + 2] * ef.z + w1es[h * 4 + 3] * ef.w;
                acc += fmaxf(hv, 0.f) * ew2s[h];
            }
            const float phys = acc + eb2v;
            const float* ap = apri + ((size_t)bt * Nn * Nn + idx) * 5;
            float pr = ap[0] * wfs[0] + ap[1] * wfs[1] + ap[2] * wfs[2]
                     + ap[3] * wfs[3] + ap[4] * wfs[4];
            pr = (pr != pr) ? 0.f : pr;      // nan_to_num
            pr = fmaxf(pr, 0.f);
            float l = lg[n * SL + m] + physw * phys + priorw * __logf(pr + 1e-6f);
            if (maski[n] | maski[m]) l = -1e9f;
            lg[n * SL + m] = l;
        }
    }
    __syncthreads();

    // ---- row softmax -> alpha (f16) ----
    if (tid < 64) {
        float mx = -3.4e38f;
        for (int m = 0; m < Nn; ++m) mx = fmaxf(mx, lg[tid * SL + m]);
        float s = 0.f;
        for (int m = 0; m < Nn; ++m) s += __expf(lg[tid * SL + m] - mx);
        const float inv = 1.f / s;
        for (int m = 0; m < Nn; ++m)
            ah[tid * SVT + m] = (_Float16)(__expf(lg[tid * SL + m] - mx) * inv);
    }
    __syncthreads();

    // ---- spatial = alpha @ V (K = 64, V transposed in LDS) ----
    gemm_f16out<2, false>(ah, SVT, vT, SVT, sph, SH, w, lane);
    __syncthreads();

    // ---- outf = spatial @ Theta^T -> rb (f32) ----
    stage_w(Theta, wbuf, tid);
    __syncthreads();
    gemm_f32out(sph, SH, wbuf, SH, rb, SR, w, lane);
    __syncthreads();

    // ---- residual (fp32 Z from global) + LayerNorm + masked transposed store ----
    {
        const int n = tid >> 2, seg = tid & 3;
        const float* xrow = x + (((size_t)(bi * Nn + n) * Tt) + ti) * Dd;
        float s = 0.f, s2 = 0.f;
#pragma unroll
        for (int i = 0; i < 8; ++i) {
            const int d = seg * 32 + i * 4;
            const float4 xv = *(const float4*)(xrow + d);
            float4 rv = *(float4*)(rb + n * SR + d);
            rv.x += xv.x; rv.y += xv.y; rv.z += xv.z; rv.w += xv.w;
            *(float4*)(rb + n * SR + d) = rv;
            s  += rv.x + rv.y + rv.z + rv.w;
            s2 += rv.x * rv.x + rv.y * rv.y + rv.z * rv.z + rv.w * rv.w;
        }
        part[tid * 2 + 0] = s;
        part[tid * 2 + 1] = s2;
    }
    __syncthreads();
    if (tid < 64) {
        float su = 0.f, sq = 0.f;
#pragma unroll
        for (int seg = 0; seg < 4; ++seg) {
            su += part[(tid * 4 + seg) * 2 + 0];
            sq += part[(tid * 4 + seg) * 2 + 1];
        }
        const float mu = su * (1.f / 128.f);
        const float var = sq * (1.f / 128.f) - mu * mu;
        stats[tid * 2 + 0] = mu;
        stats[tid * 2 + 1] = rsqrtf(var + 1e-5f);
    }
    __syncthreads();
    {
        const int n = tid >> 2, seg = tid & 3;
        const float mu = stats[n * 2 + 0], rstd = stats[n * 2 + 1];
        const int msk = maski[n];
        float* orow = out + (((size_t)(bi * Nn + n) * Tt) + ti) * Dd;
#pragma unroll
        for (int i = 0; i < 8; ++i) {
            const int d = seg * 32 + i * 4;
            const float4 rv = *(const float4*)(rb + n * SR + d);
            float4 o;
            o.x = msk ? 0.f : (rv.x - mu) * rstd * gs[d + 0] + bs[d + 0];
            o.y = msk ? 0.f : (rv.y - mu) * rstd * gs[d + 1] + bs[d + 1];
            o.z = msk ? 0.f : (rv.z - mu) * rstd * gs[d + 2] + bs[d + 2];
            o.w = msk ? 0.f : (rv.w - mu) * rstd * gs[d + 3] + bs[d + 3];
            *(float4*)(orow + d) = o;
        }
    }
}

extern "C" void kernel_launch(void* const* d_in, const int* in_sizes, int n_in,
                              void* d_out, int out_size, void* d_ws, size_t ws_size,
                              hipStream_t stream) {
    const float* x      = (const float*)d_in[0];
    const float* edgef  = (const float*)d_in[1];
    const float* apri   = (const float*)d_in[2];
    const unsigned char* mask = (const unsigned char*)d_in[3];
    const float* Wq     = (const float*)d_in[4];
    const float* Wk     = (const float*)d_in[5];
    const float* Wv     = (const float*)d_in[6];
    const float* Theta  = (const float*)d_in[7];
    const float* Wfuse  = (const float*)d_in[8];
    const float* eW1    = (const float*)d_in[9];
    const float* eb1    = (const float*)d_in[10];
    const float* eW2    = (const float*)d_in[11];
    const float* eb2    = (const float*)d_in[12];
    const float* lng    = (const float*)d_in[13];
    const float* lnb    = (const float*)d_in[14];
    const float* physw  = (const float*)d_in[15];
    const float* priorw = (const float*)d_in[16];
    float* out = (float*)d_out;

    dyn_spatial_gnn_kernel<<<dim3(128), dim3(256), SMEM_BYTES, stream>>>(
        x, edgef, apri, mask, Wq, Wk, Wv, Theta, Wfuse, eW1, eb1, eW2, eb2,
        lng, lnb, physw, priorw, out);
}
